// MV_CLIP_10539849744948
// MI455X (gfx1250) — matrix operand order repaired and checked by reading, not compiled, not run on hardware
//
#include <hip/hip_runtime.h>
#include <math.h>

// Problem constants (from reference)
#define BB    64
#define LT_   512
#define LI_   1024
#define DD    512
#define DI_   768
#define KTOP  5
#define NLAB_ 2

typedef unsigned short u16;
typedef __attribute__((ext_vector_type(16))) __bf16 v16bf;
typedef __attribute__((ext_vector_type(8)))  float  v8f;

union Frag16 { uint4 u[2]; v16bf v; };

// ---------------- reductions ----------------
__device__ __forceinline__ float wave_red_sum(float v) {
#pragma unroll
  for (int o = 16; o > 0; o >>= 1) v += __shfl_xor(v, o);
  return v;
}

__device__ __forceinline__ float block_red_sum256(float v, float* sr) {
  int tid = threadIdx.x;
  sr[tid] = v; __syncthreads();
#pragma unroll
  for (int s = 128; s > 0; s >>= 1) { if (tid < s) sr[tid] += sr[tid + s]; __syncthreads(); }
  float r = sr[0]; __syncthreads();
  return r;
}

__device__ __forceinline__ float gelu_exact(float x) {
  return 0.5f * x * (1.0f + erff(x * 0.70710678118654752f));
}

__device__ __forceinline__ u16 f2bf(float f) {
  unsigned u = __float_as_uint(f);
  return (u16)((u + 0x7FFFu + ((u >> 16) & 1u)) >> 16);
}

// ---------------- prep: fp32 -> bf16 ----------------
__global__ void cvt_bf16_kernel(const float* __restrict__ in, u16* __restrict__ out, size_t n) {
  size_t i = (size_t)blockIdx.x * blockDim.x + threadIdx.x;
  size_t stride = (size_t)gridDim.x * blockDim.x;
  for (; i < n; i += stride) out[i] = f2bf(in[i]);
}

// Wt[n][k] = W[k][n], bf16 (column-major weight for WMMA B fragments)
__global__ void transpose_w_bf16_kernel(const float* __restrict__ W, u16* __restrict__ Wt) {
  int n = blockIdx.x;
  for (int k = threadIdx.x; k < DD; k += blockDim.x)
    Wt[(size_t)n * DD + k] = f2bf(W[(size_t)k * DD + n]);
}

// ---------------- query rows: proj of text[b,last] and image[b,0] ----------------
__device__ void proj_one_row(const float* __restrict__ xrow, const float* __restrict__ W,
                             const float* __restrict__ g, const float* __restrict__ be,
                             float* __restrict__ qout, float* sX, float* sr) {
  int tid = threadIdx.x;
  for (int k = tid; k < DD; k += 256) sX[k] = xrow[k];
  __syncthreads();
  int n0 = tid, n1 = tid + 256;
  float a0 = 0.f, a1 = 0.f;
  for (int k = 0; k < DD; ++k) {
    float x = sX[k];
    a0 += x * W[(size_t)k * DD + n0];
    a1 += x * W[(size_t)k * DD + n1];
  }
  float ss = block_red_sum256(a0 * a0 + a1 * a1, sr);
  float inv = 1.f / fmaxf(sqrtf(ss), 1e-12f);
  a0 *= inv; a1 *= inv;
  float mean = block_red_sum256(a0 + a1, sr) * (1.f / 512.f);
  float var = block_red_sum256((a0 - mean) * (a0 - mean) + (a1 - mean) * (a1 - mean), sr) * (1.f / 512.f);
  float rstd = rsqrtf(var + 1e-5f);
  qout[n0] = (a0 - mean) * rstd * g[n0] + be[n0];
  qout[n1] = (a1 - mean) * rstd * g[n1] + be[n1];
  __syncthreads();
}

__global__ void qrows_kernel(const float* __restrict__ tE, const float* __restrict__ iE,
                             const int* __restrict__ mask,
                             const float* __restrict__ Wt, const float* __restrict__ Wi,
                             const float* __restrict__ tg, const float* __restrict__ tb2,
                             const float* __restrict__ ig, const float* __restrict__ ib2,
                             float* __restrict__ q_t, float* __restrict__ q_i,
                             int* __restrict__ last_idx) {
  __shared__ float sX[512];
  __shared__ float sr[256];
  int b = blockIdx.x, tid = threadIdx.x;
  float msum = 0.f;
  for (int t = tid; t < LT_; t += 256) msum += (float)mask[(size_t)b * LT_ + t];
  int last = (int)(block_red_sum256(msum, sr) + 0.5f) - 1;
  if (last < 0) last = 0;
  if (tid == 0) last_idx[b] = last;
  proj_one_row(tE + ((size_t)b * LT_ + last) * DD, Wt, tg, tb2, q_t + (size_t)b * DD, sX, sr);
  proj_one_row(iE + ((size_t)b * LI_) * DD, Wi, ig, ib2, q_i + (size_t)b * DD, sX, sr);
}

// ---------------- WMMA: fused  proj-GEMM + l2norm + LN + dot(q)  ----------------
// grid = (Lrows/16, B). Each block: 16 rows of embeds (bf16).
// A tile (16x512 bf16) is staged into LDS via global_load_async_to_lds_b128 (ASYNCcnt),
// shared by all 8 waves; B (transposed bf16 weights, L2-resident) streams from VMEM.
// 16x512 fp32 result tile goes to LDS, then fused per-row l2norm+LN+dot(q[b]).
#define APAD 520  // row stride in u16: 1040B = 260 words; 260 % 64 = 4 -> conflict-free ds_b128
__global__ void __launch_bounds__(256)
projdot_kernel(const u16* __restrict__ Abf, const u16* __restrict__ Wt,
               const float* __restrict__ q, const float* __restrict__ lng,
               const float* __restrict__ lnb, const float* __restrict__ logit_scale,
               float* __restrict__ outvals, int Lrows) {
  __shared__ u16  sA[16][APAD];
  __shared__ float tile[16][520];
  int tid = threadIdx.x;
  int wid = tid >> 5, lane = tid & 31;
  int half = lane >> 4, lr = lane & 15;
  int b = blockIdx.y;
  int rloc = blockIdx.x * 16;
  size_t row0 = (size_t)b * Lrows + rloc;

  // ---- async stage A tile: 16 rows x 512 bf16 = 1024 x 16B chunks, 4 per thread ----
#pragma unroll
  for (int j = 0; j < 4; ++j) {
    int chunk = tid + 256 * j;          // 0..1023
    int row = chunk >> 6;               // 64 chunks of 8 u16 per row
    int col8 = (chunk & 63) * 8;
    unsigned lds_off = (unsigned)(uintptr_t)&sA[row][col8];
    unsigned long long ga = (unsigned long long)(uintptr_t)(Abf + (row0 + row) * DD + col8);
    asm volatile("global_load_async_to_lds_b128 %0, %1, off"
                 :: "v"(lds_off), "v"(ga) : "memory");
  }
  asm volatile("s_wait_asynccnt 0x0" ::: "memory");
  __syncthreads();

  v8f acc[4] = {};
  int n0 = wid * 64;

#pragma unroll 1
  for (int ks = 0; ks < 16; ++ks) {
    int k0 = ks * 32;
    Frag16 af;
    const u16* As = &sA[lr][k0 + half * 8];
    af.u[0] = *(const uint4*)(As);        // ds_load_b128
    af.u[1] = *(const uint4*)(As + 16);
#pragma unroll
    for (int tn = 0; tn < 4; ++tn) {
      int n = n0 + tn * 16 + lr;
      const u16* Bb = Wt + (size_t)n * DD + k0 + half * 8;
      Frag16 bf;
      bf.u[0] = *(const uint4*)(Bb);
      bf.u[1] = *(const uint4*)(Bb + 16);
      if (tn == 0) __builtin_prefetch(Bb + 32, 0, 1);  // next k-chunk of this B column
      acc[tn] = __builtin_amdgcn_wmma_f32_16x16x32_bf16(
          false, af.v, false, bf.v, (short)0, acc[tn], false, false);
    }
  }
#pragma unroll
  for (int tn = 0; tn < 4; ++tn) {
    int n = n0 + tn * 16 + lr;
#pragma unroll
    for (int r = 0; r < 8; ++r) tile[r + 8 * half][n] = acc[tn][r];
  }
  __syncthreads();

  float expls = __expf(logit_scale[0]);
#pragma unroll 1
  for (int rr = 0; rr < 2; ++rr) {
    int m = wid * 2 + rr;
    float x[16];
    float ss = 0.f;
#pragma unroll
    for (int j = 0; j < 16; ++j) { x[j] = tile[m][lane + 32 * j]; ss += x[j] * x[j]; }
    ss = wave_red_sum(ss);
    float inv = 1.f / fmaxf(sqrtf(ss), 1e-12f);
    float sm = 0.f;
#pragma unroll
    for (int j = 0; j < 16; ++j) { x[j] *= inv; sm += x[j]; }
    sm = wave_red_sum(sm) * (1.f / 512.f);
    float vv = 0.f;
#pragma unroll
    for (int j = 0; j < 16; ++j) { float d = x[j] - sm; vv += d * d; }
    vv = wave_red_sum(vv) * (1.f / 512.f);
    float rstd = rsqrtf(vv + 1e-5f);
    float dot = 0.f;
#pragma unroll
    for (int j = 0; j < 16; ++j) {
      int idx = lane + 32 * j;
      float y = (x[j] - sm) * rstd * lng[idx] + lnb[idx];
      dot += y * q[(size_t)b * DD + idx];
    }
    dot = wave_red_sum(dot);
    if (lane == 0) outvals[(size_t)b * Lrows + rloc + m] = dot * expls;
  }
}

// ---------------- softmax + context ----------------
__global__ void ctx_kernel(const float* __restrict__ vals, const float* __restrict__ emb,
                           float* __restrict__ outc, int L) {
  __shared__ float sv[1024];
  __shared__ float sr[256];
  int b = blockIdx.x, tid = threadIdx.x;
  const float* v = vals + (size_t)b * L;
  float lmax = -1e30f;
  for (int i = tid; i < L; i += 256) { float t = v[i]; sv[i] = t; lmax = fmaxf(lmax, t); }
  sr[tid] = lmax; __syncthreads();
#pragma unroll
  for (int s = 128; s > 0; s >>= 1) { if (tid < s) sr[tid] = fmaxf(sr[tid], sr[tid + s]); __syncthreads(); }
  float mx = sr[0]; __syncthreads();
  float lsum = 0.f;
  for (int i = tid; i < L; i += 256) { float e = __expf(sv[i] - mx); sv[i] = e; lsum += e; }
  __syncthreads();
  float invt = 1.f / block_red_sum256(lsum, sr);
  const float* eb = emb + (size_t)b * L * DD;
  float a0 = 0.f, a1 = 0.f;
  for (int i = 0; i < L; ++i) {
    float p = sv[i] * invt;
    a0 += p * eb[(size_t)i * DD + tid];
    a1 += p * eb[(size_t)i * DD + tid + 256];
  }
  outc[(size_t)b * DD + tid] = a0;
  outc[(size_t)b * DD + tid + 256] = a1;
}

// ---------------- top-5 + value-gemm + capsule routing ----------------
__global__ void topk_route_kernel(const float* __restrict__ vals, const float* __restrict__ emb,
                                  const float* __restrict__ Wv, float* __restrict__ vout, int L) {
  __shared__ float sv[1024];
  __shared__ float su[KTOP][512];
  __shared__ float sx[512];
  __shared__ float svv[512];
  __shared__ float sr[256];
  __shared__ int   sidx[256];
  __shared__ int   topi[KTOP];
  __shared__ float topv[KTOP];
  int b = blockIdx.x, tid = threadIdx.x;
  const float* v = vals + (size_t)b * L;
  for (int i = tid; i < L; i += 256) sv[i] = v[i];
  __syncthreads();
  for (int k = 0; k < KTOP; ++k) {
    float bm = -1e30f; int bi = 0;
    for (int i = tid; i < L; i += 256) if (sv[i] > bm) { bm = sv[i]; bi = i; }
    sr[tid] = bm; sidx[tid] = bi; __syncthreads();
#pragma unroll
    for (int s = 128; s > 0; s >>= 1) {
      if (tid < s && sr[tid + s] > sr[tid]) { sr[tid] = sr[tid + s]; sidx[tid] = sidx[tid + s]; }
      __syncthreads();
    }
    if (tid == 0) { topi[k] = sidx[0]; topv[k] = sr[0]; sv[sidx[0]] = -1e30f; }
    __syncthreads();
  }
  // u[k] = emb[b, topi[k]] @ Wv   (Wv: [512][512] row-major)
  for (int k = 0; k < KTOP; ++k) {
    const float* xr = emb + ((size_t)b * L + topi[k]) * DD;
    for (int j = tid; j < DD; j += 256) sx[j] = xr[j];
    __syncthreads();
#pragma unroll
    for (int j = 0; j < 2; ++j) {
      int n = tid + 256 * j;
      float a = 0.f;
      for (int kk = 0; kk < DD; ++kk) a += sx[kk] * Wv[(size_t)kk * DD + n];
      su[k][n] = a;
    }
    __syncthreads();
  }
  // dynamic routing, 3 iters
  float bl[KTOP];
#pragma unroll
  for (int k = 0; k < KTOP; ++k) bl[k] = topv[k];
  for (int it = 0; it < 3; ++it) {
    float m = bl[0];
#pragma unroll
    for (int k = 1; k < KTOP; ++k) m = fmaxf(m, bl[k]);
    float es[KTOP], tot = 0.f;
#pragma unroll
    for (int k = 0; k < KTOP; ++k) { es[k] = __expf(bl[k] - m); tot += es[k]; }
    float invt = 1.f / tot;
    float s0 = 0.f, s1 = 0.f;
#pragma unroll
    for (int k = 0; k < KTOP; ++k) {
      float c = es[k] * invt;
      s0 += c * su[k][tid];
      s1 += c * su[k][tid + 256];
    }
    float sq = block_red_sum256(s0 * s0 + s1 * s1, sr);
    float scale = (sq / (1.f + sq)) / sqrtf(sq + 1e-8f);
    svv[tid] = s0 * scale; svv[tid + 256] = s1 * scale;
    __syncthreads();
    if (it < 2) {
      for (int k = 0; k < KTOP; ++k) {
        float p = su[k][tid] * svv[tid] + su[k][tid + 256] * svv[tid + 256];
        bl[k] += block_red_sum256(p, sr);
      }
    }
  }
  vout[(size_t)b * DD + tid] = svv[tid];
  vout[(size_t)b * DD + tid + 256] = svv[tid + 256];
}

// ---------------- per-row FFN + residual + LN ----------------
__global__ void ffn_kernel(const float* __restrict__ vrow, const float* __restrict__ emb,
                           const int* __restrict__ last_idx, int use_last,
                           const float* __restrict__ W1, const float* __restrict__ b1,
                           const float* __restrict__ W2, const float* __restrict__ b2,
                           const float* __restrict__ g, const float* __restrict__ be,
                           float* __restrict__ fimout, int L) {
  __shared__ float sx[512];
  __shared__ float sh[2048];
  __shared__ float sr[256];
  int b = blockIdx.x, tid = threadIdx.x;
  for (int j = tid; j < DD; j += 256) sx[j] = vrow[(size_t)b * DD + j];
  __syncthreads();
#pragma unroll 1
  for (int j = 0; j < 8; ++j) {
    int n = tid + 256 * j;
    float a = b1[n];
    for (int k = 0; k < DD; ++k) a += sx[k] * W1[(size_t)k * 2048 + n];
    sh[n] = gelu_exact(a);
  }
  __syncthreads();
  int row = use_last ? last_idx[b] : 0;
  const float* res = emb + ((size_t)b * L + row) * DD;
  float y0, y1;
  {
    int n = tid;
    float a = b2[n];
    for (int k = 0; k < 2048; ++k) a += sh[k] * W2[(size_t)k * DD + n];
    y0 = a + res[n];
  }
  {
    int n = tid + 256;
    float a = b2[n];
    for (int k = 0; k < 2048; ++k) a += sh[k] * W2[(size_t)k * DD + n];
    y1 = a + res[n];
  }
  float sm = block_red_sum256(y0 + y1, sr) * (1.f / 512.f);
  float vr = block_red_sum256((y0 - sm) * (y0 - sm) + (y1 - sm) * (y1 - sm), sr) * (1.f / 512.f);
  float rstd = rsqrtf(vr + 1e-5f);
  fimout[(size_t)b * DD + tid] = (y0 - sm) * rstd * g[tid] + be[tid];
  fimout[(size_t)b * DD + tid + 256] = (y1 - sm) * rstd * g[tid + 256] + be[tid + 256];
}

// ---------------- fuse + classifier heads + CE ----------------
__device__ void fuse2(const float* t, const float* i_, const float* __restrict__ att_w,
                      float* out, float* sr, int tid) {
  float pt = 0.f, pi = 0.f;
#pragma unroll
  for (int j = 0; j < 2; ++j) {
    int n = tid + 256 * j;
    pt += t[n] * att_w[n];
    pi += i_[n] * att_w[n];
  }
  float tw = block_red_sum256(pt, sr);
  float iw = block_red_sum256(pi, sr);
  float m = fmaxf(tw, iw);
  float e0 = __expf(tw - m), e1 = __expf(iw - m);
  float inv = 1.f / (e0 + e1);
  float a0 = e0 * inv, a1 = e1 * inv;
#pragma unroll
  for (int j = 0; j < 2; ++j) {
    int n = tid + 256 * j;
    out[n] = a0 * t[n] + a1 * i_[n];
  }
  __syncthreads();
}

__global__ void head_kernel(const float* __restrict__ tfeat_in, const float* __restrict__ ifeat_in,
                            const float* __restrict__ txtw, const float* __restrict__ txtb,
                            const float* __restrict__ imgw, const float* __restrict__ imgb,
                            const float* __restrict__ cls_t_w, const float* __restrict__ cls_t_b,
                            const float* __restrict__ cls_i_w, const float* __restrict__ cls_i_b,
                            const float* __restrict__ text_c, const float* __restrict__ image_c,
                            const float* __restrict__ fim_t, const float* __restrict__ fim_i,
                            const float* __restrict__ att_w,
                            const float* __restrict__ extra_w, const float* __restrict__ extra_b,
                            const float* __restrict__ cls_f_w, const float* __restrict__ cls_f_b,
                            const int* __restrict__ labels,
                            float* __restrict__ score_out, float* __restrict__ ce_part) {
  __shared__ float sA[768];
  __shared__ float sF[768];
  __shared__ float sT[512], sI[512];
  __shared__ float sBF[512], sFF[512], sFU[512];
  __shared__ float sr[256];
  int b = blockIdx.x, tid = threadIdx.x;

  // text head: gelu(text_feature @ txt_lin_w + b)
  for (int j = tid; j < DD; j += 256) sA[j] = tfeat_in[(size_t)b * DD + j];
  __syncthreads();
#pragma unroll 1
  for (int j = 0; j < 2; ++j) {
    int n = tid + 256 * j;
    float a = txtb[n];
    for (int k = 0; k < DD; ++k) a += sA[k] * txtw[(size_t)k * DD + n];
    sF[n] = gelu_exact(a);
  }
  __syncthreads();
  float p0 = 0.f, p1 = 0.f;
#pragma unroll
  for (int j = 0; j < 2; ++j) {
    int n = tid + 256 * j;
    p0 += sF[n] * cls_t_w[n * 2 + 0];
    p1 += sF[n] * cls_t_w[n * 2 + 1];
  }
  float lt0 = block_red_sum256(p0, sr) + cls_t_b[0];
  float lt1 = block_red_sum256(p1, sr) + cls_t_b[1];

  // image head (768)
  for (int j = tid; j < DI_; j += 256) sA[j] = ifeat_in[(size_t)b * DI_ + j];
  __syncthreads();
#pragma unroll 1
  for (int j = 0; j < 3; ++j) {
    int n = tid + 256 * j;
    float a = imgb[n];
    for (int k = 0; k < DI_; ++k) a += sA[k] * imgw[(size_t)k * DI_ + n];
    sF[n] = gelu_exact(a);
  }
  __syncthreads();
  p0 = 0.f; p1 = 0.f;
#pragma unroll
  for (int j = 0; j < 3; ++j) {
    int n = tid + 256 * j;
    p0 += sF[n] * cls_i_w[n * 2 + 0];
    p1 += sF[n] * cls_i_w[n * 2 + 1];
  }
  float li0 = block_red_sum256(p0, sr) + cls_i_b[0];
  float li1 = block_red_sum256(p1, sr) + cls_i_b[1];

  // base fuse
  for (int j = 0; j < 2; ++j) {
    int n = tid + 256 * j;
    sT[n] = text_c[(size_t)b * DD + n];
    sI[n] = image_c[(size_t)b * DD + n];
  }
  __syncthreads();
  fuse2(sT, sI, att_w, sBF, sr, tid);
  // fim fuse
  for (int j = 0; j < 2; ++j) {
    int n = tid + 256 * j;
    sT[n] = fim_t[(size_t)b * DD + n];
    sI[n] = fim_i[(size_t)b * DD + n];
  }
  __syncthreads();
  fuse2(sT, sI, att_w, sFF, sr, tid);

  // fuse_feature = concat(base, fim) @ extra_w + extra_b
#pragma unroll 1
  for (int j = 0; j < 2; ++j) {
    int n = tid + 256 * j;
    float a = extra_b[n];
    for (int k = 0; k < DD; ++k) a += sBF[k] * extra_w[(size_t)k * DD + n];
    for (int k = 0; k < DD; ++k) a += sFF[k] * extra_w[(size_t)(k + DD) * DD + n];
    sFU[n] = a;
  }
  __syncthreads();
  p0 = 0.f; p1 = 0.f;
#pragma unroll
  for (int j = 0; j < 2; ++j) {
    int n = tid + 256 * j;
    p0 += sFU[n] * cls_f_w[n * 2 + 0];
    p1 += sFU[n] * cls_f_w[n * 2 + 1];
  }
  float lf0 = block_red_sum256(p0, sr) + cls_f_b[0];
  float lf1 = block_red_sum256(p1, sr) + cls_f_b[1];

  if (tid == 0) {
    float m, e0, e1, inv;
    m = fmaxf(lf0, lf1); e0 = __expf(lf0 - m); e1 = __expf(lf1 - m); inv = 1.f / (e0 + e1);
    float f0 = e0 * inv, f1 = e1 * inv;
    float lsef = m + logf(e0 + e1);
    m = fmaxf(lt0, lt1); e0 = __expf(lt0 - m); e1 = __expf(lt1 - m); inv = 1.f / (e0 + e1);
    float t0 = e0 * inv, t1 = e1 * inv;
    float lset = m + logf(e0 + e1);
    m = fmaxf(li0, li1); e0 = __expf(li0 - m); e1 = __expf(li1 - m); inv = 1.f / (e0 + e1);
    float i0 = e0 * inv, i1 = e1 * inv;
    float lsei = m + logf(e0 + e1);
    score_out[b * 2 + 0] = f0 + t0 + i0;
    score_out[b * 2 + 1] = f1 + t1 + i1;
    int lab = labels[b];
    float ce = (lsef - (lab == 0 ? lf0 : lf1))
             + (lset - (lab == 0 ? lt0 : lt1))
             + (lsei - (lab == 0 ? li0 : li1));
    ce_part[b] = ce;
  }
}

__global__ void finalize_kernel(const float* __restrict__ ce_part, float* __restrict__ out) {
  __shared__ float sr[64];
  int tid = threadIdx.x;
  sr[tid] = ce_part[tid]; __syncthreads();
#pragma unroll
  for (int s = 32; s > 0; s >>= 1) { if (tid < s) sr[tid] += sr[tid + s]; __syncthreads(); }
  if (tid == 0) out[0] = sr[0] * (1.f / 64.f);
}

// ---------------- host orchestration ----------------
extern "C" void kernel_launch(void* const* d_in, const int* in_sizes, int n_in,
                              void* d_out, int out_size, void* d_ws, size_t ws_size,
                              hipStream_t stream) {
  (void)in_sizes; (void)n_in; (void)out_size; (void)ws_size;
  // setup_inputs() dict order, params flattened in dict insertion order.
  const float* text_embeds  = (const float*)d_in[0];
  const float* image_embeds = (const float*)d_in[1];
  const float* text_feature = (const float*)d_in[2];
  const float* image_feature= (const float*)d_in[3];
  const int*   attn_mask    = (const int*)d_in[4];
  const int*   labels       = (const int*)d_in[5];
  const float* txt_lin_w = (const float*)d_in[6];
  const float* txt_lin_b = (const float*)d_in[7];
  const float* img_lin_w = (const float*)d_in[8];
  const float* img_lin_b = (const float*)d_in[9];
  const float* cim_t_w   = (const float*)d_in[10];
  const float* cim_i_w   = (const float*)d_in[11];
  const float* cim_t_ln_g= (const float*)d_in[12];
  const float* cim_t_ln_b= (const float*)d_in[13];
  const float* cim_i_ln_g= (const float*)d_in[14];
  const float* cim_i_ln_b= (const float*)d_in[15];
  const float* logit_scale=(const float*)d_in[16];
  const float* dynrt_img_w=(const float*)d_in[17];
  const float* dynrt_txt_w=(const float*)d_in[18];
  const float* fim_t_w1  = (const float*)d_in[19];
  const float* fim_t_b1  = (const float*)d_in[20];
  const float* fim_t_w2  = (const float*)d_in[21];
  const float* fim_t_b2  = (const float*)d_in[22];
  const float* fim_i_w1  = (const float*)d_in[23];
  const float* fim_i_b1  = (const float*)d_in[24];
  const float* fim_i_w2  = (const float*)d_in[25];
  const float* fim_i_b2  = (const float*)d_in[26];
  const float* fim_t_ln_g= (const float*)d_in[27];
  const float* fim_t_ln_b= (const float*)d_in[28];
  const float* fim_i_ln_g= (const float*)d_in[29];
  const float* fim_i_ln_b= (const float*)d_in[30];
  const float* extra_w   = (const float*)d_in[31];
  const float* extra_b   = (const float*)d_in[32];
  const float* cls_fuse_w= (const float*)d_in[33];
  const float* cls_fuse_b= (const float*)d_in[34];
  const float* cls_text_w= (const float*)d_in[35];
  const float* cls_text_b= (const float*)d_in[36];
  const float* cls_img_w = (const float*)d_in[37];
  const float* cls_img_b = (const float*)d_in[38];
  const float* att_w     = (const float*)d_in[39];

  float* out = (float*)d_out;            // [0]=loss, [1..128]=score

  // workspace carve-up
  char* wp = (char*)d_ws;
  auto alloc = [&](size_t bytes) -> void* {
    void* p = (void*)wp;
    wp += (bytes + 255) & ~(size_t)255;
    return p;
  };
  const size_t nTE = (size_t)BB * LT_ * DD;
  const size_t nIE = (size_t)BB * LI_ * DD;
  u16*  tE_bf  = (u16*)alloc(nTE * sizeof(u16));
  u16*  iE_bf  = (u16*)alloc(nIE * sizeof(u16));
  u16*  WtT    = (u16*)alloc((size_t)DD * DD * sizeof(u16));
  u16*  WiT    = (u16*)alloc((size_t)DD * DD * sizeof(u16));
  float* q_t   = (float*)alloc((size_t)BB * DD * sizeof(float));
  float* q_i   = (float*)alloc((size_t)BB * DD * sizeof(float));
  int*   last_idx = (int*)alloc((size_t)BB * sizeof(int));
  float* rowvals = (float*)alloc((size_t)BB * LI_ * sizeof(float));
  float* colvals = (float*)alloc((size_t)BB * LT_ * sizeof(float));
  float* text_c  = (float*)alloc((size_t)BB * DD * sizeof(float));
  float* image_c = (float*)alloc((size_t)BB * DD * sizeof(float));
  float* v_t     = (float*)alloc((size_t)BB * DD * sizeof(float));
  float* v_i     = (float*)alloc((size_t)BB * DD * sizeof(float));
  float* fim_t   = (float*)alloc((size_t)BB * DD * sizeof(float));
  float* fim_i   = (float*)alloc((size_t)BB * DD * sizeof(float));
  float* ce_part = (float*)alloc((size_t)BB * sizeof(float));

  // 1) bf16 conversions + weight transposes
  cvt_bf16_kernel<<<8192, 256, 0, stream>>>(text_embeds, tE_bf, nTE);
  cvt_bf16_kernel<<<8192, 256, 0, stream>>>(image_embeds, iE_bf, nIE);
  transpose_w_bf16_kernel<<<DD, 256, 0, stream>>>(cim_t_w, WtT);
  transpose_w_bf16_kernel<<<DD, 256, 0, stream>>>(cim_i_w, WiT);

  // 2) query projection rows (text[b,last], image[b,0])
  qrows_kernel<<<BB, 256, 0, stream>>>(text_embeds, image_embeds, attn_mask,
                                       cim_t_w, cim_i_w,
                                       cim_t_ln_g, cim_t_ln_b, cim_i_ln_g, cim_i_ln_b,
                                       q_t, q_i, last_idx);

  // 3) fused WMMA proj + norm + dot:
  //    rowvals[b,i] = <image_proj[b,i], q_t[b]> * e^ls ; colvals[b,t] = <text_proj[b,t], q_i[b]> * e^ls
  projdot_kernel<<<dim3(LI_ / 16, BB), 256, 0, stream>>>(iE_bf, WiT, q_t, cim_i_ln_g, cim_i_ln_b,
                                                         logit_scale, rowvals, LI_);
  projdot_kernel<<<dim3(LT_ / 16, BB), 256, 0, stream>>>(tE_bf, WtT, q_i, cim_t_ln_g, cim_t_ln_b,
                                                         logit_scale, colvals, LT_);

  // 4) softmax-context rows
  ctx_kernel<<<BB, 256, 0, stream>>>(rowvals, image_embeds, text_c, LI_);
  ctx_kernel<<<BB, 256, 0, stream>>>(colvals, text_embeds, image_c, LT_);

  // 5) top-5 + value projection + capsule routing
  topk_route_kernel<<<BB, 256, 0, stream>>>(rowvals, image_embeds, dynrt_img_w, v_t, LI_);
  topk_route_kernel<<<BB, 256, 0, stream>>>(colvals, text_embeds, dynrt_txt_w, v_i, LT_);

  // 6) per-row FFN + residual + LN
  ffn_kernel<<<BB, 256, 0, stream>>>(v_t, text_embeds, last_idx, 1,
                                     fim_t_w1, fim_t_b1, fim_t_w2, fim_t_b2,
                                     fim_t_ln_g, fim_t_ln_b, fim_t, LT_);
  ffn_kernel<<<BB, 256, 0, stream>>>(v_i, image_embeds, last_idx, 0,
                                     fim_i_w1, fim_i_b1, fim_i_w2, fim_i_b2,
                                     fim_i_ln_g, fim_i_ln_b, fim_i, LI_);

  // 7) fuse + heads + CE partials, then reduce loss
  head_kernel<<<BB, 256, 0, stream>>>(text_feature, image_feature,
                                      txt_lin_w, txt_lin_b, img_lin_w, img_lin_b,
                                      cls_text_w, cls_text_b, cls_img_w, cls_img_b,
                                      text_c, image_c, fim_t, fim_i, att_w,
                                      extra_w, extra_b, cls_fuse_w, cls_fuse_b,
                                      labels, out + 1, ce_part);
  finalize_kernel<<<1, 64, 0, stream>>>(ce_part, out);
}
